// KalmanNet2_29300266893514
// MI455X (gfx1250) — matrix operand order repaired and checked by reading, not compiled, not run on hardware
//
#include <hip/hip_runtime.h>
#include <hip/hip_bf16.h>
#include <stddef.h>

// ---------------------------------------------------------------------------
// KalmanNet arch-2 on gfx1250: 16 WGs (16 batch rows each) x 256 threads
// (8 wave32). All linears/GRU matmuls via v_wmma_f32_16x16x32_bf16 with fp32
// accumulation; recurrent carries kept in fp32 LDS. Weights pre-swizzled to
// bf16 B-fragment layout in d_ws; biases packed in d_ws and staged to LDS.
// ---------------------------------------------------------------------------

typedef __bf16 v16bf __attribute__((ext_vector_type(16)));
typedef float  v8f   __attribute__((ext_vector_type(8)));

#define WAVES 8

// swizzled-weight offsets in bf16 elements (K padded to 32, N to 16)
#define OFF_Q_IH    0u        // 160x192
#define OFF_Q_HH    30720u    // 64x192
#define OFF_SIG_IH  43008u    // 224x192
#define OFF_SIG_HH  86016u    // 64x192
#define OFF_S_IH    98304u    // 384x192
#define OFF_S_HH    172032u   // 64x192
#define OFF_FC1     184320u   // 64x64
#define OFF_FC2A    188416u   // 128x5120
#define OFF_FC2B    843776u   // 5120x64
#define OFF_FC3     1171456u  // 128x64
#define OFF_FC4     1179648u  // 128x64
#define OFF_FC5     1187840u  // 32x160
#define OFF_FC6     1192960u  // 32x160
#define OFF_FC7     1198080u  // 32x320
#define SW_TOTAL    1208320u

// packed-bias offsets (float elements) within the bias block
#define B_Q_IH   0
#define B_Q_HH   192
#define B_SIG_IH 384
#define B_SIG_HH 576
#define B_S_IH   768
#define B_S_HH   960
#define B_FC1    1152
#define B_FC2A   1216
#define B_FC2B   6336
#define B_FC3    6400
#define B_FC4    6464
#define B_FC5    6528
#define B_FC6    6688
#define B_FC7    6848
#define B_TOTAL  7168

#define WS_BIAS_BYTE_OFF ((size_t)SW_TOTAL * 2)   // bf16 weights first
#define WS_TOTAL_BYTES   (WS_BIAS_BYTE_OFF + (size_t)B_TOTAL * 4)

struct KParams {
  const float* y;
  const float* biases;   // packed bias block in d_ws
  const float *F, *H;
  const __bf16* sw;
  float* out;
};

// ---------------------------- device helpers -------------------------------

__device__ inline v8f zero8() {
  v8f c;
#pragma unroll
  for (int q = 0; q < 8; ++q) c[q] = 0.f;
  return c;
}

__device__ inline v8f bfwmma(v16bf a, v16bf b, v8f c) {
  return __builtin_amdgcn_wmma_f32_16x16x32_bf16(false, a, false, b,
                                                 (short)0, c, false, false);
}

// A-fragment gather (16x32, 16-bit): lane m=lane&15; lanes>=16 shift K by +8/+24.
template <typename T>
__device__ inline v16bf load_a(const T* buf, int stride, int kt, int lane) {
  const int m  = lane & 15;
  const int hi = (lane >> 4) & 1;
  const T* p = buf + m * stride + kt * 32 + hi * 8;
  v16bf a;
#pragma unroll
  for (int j = 0; j < 16; ++j) {
    const int k = ((j >> 3) << 4) + (j & 7);   // 0..7 then 16..23
    a[j] = (__bf16)p[k];
  }
  return a;
}

// C/D f32 16x16 scatter: VGPR r -> row (hi?8:0)+r, col lane&15.
template <typename OT>
__device__ inline void store_tile(OT* out, int ostride, int colbase, v8f c,
                                  const float* bias, bool relu, int lane) {
  const int col = lane & 15;
  const int hi  = (lane >> 4) & 1;
  const float badd = bias ? bias[col] : 0.f;
#pragma unroll
  for (int r = 0; r < 8; ++r) {
    float v = c[r] + badd;
    if (relu) v = fmaxf(v, 0.f);
    out[(hi * 8 + r) * ostride + colbase + col] = (OT)v;
  }
}

// out[16 x ntiles*16] = A[16 x ktiles*32] * Bsw (+bias)(+relu); n-tiles over waves
template <typename AT, typename OT>
__device__ inline void gemm16(const AT* abuf, int astride, const __bf16* bsw,
                              int ktiles, int ntiles, OT* out, int ostride,
                              const float* bias, bool relu, int wave, int lane) {
  for (int nt = wave; nt < ntiles; nt += WAVES) {
    v8f c = zero8();
    const __bf16* bp = bsw + (size_t)nt * ktiles * 512;
    for (int kt = 0; kt < ktiles; ++kt) {
      v16bf av = load_a(abuf, astride, kt, lane);
      v16bf bv = *(const v16bf*)(bp + (size_t)kt * 512);
      c = bfwmma(av, bv, c);
    }
    store_tile<OT>(out, ostride, nt * 16, c,
                   bias ? bias + nt * 16 : nullptr, relu, lane);
  }
}

__device__ inline float sigm(float x) { return 1.f / (1.f + __expf(-x)); }

// ------------------------- weight swizzle kernel ---------------------------
// dst tile (nt,kt): lane l, slot j holds W^T[k][n] = w[n*din + k],
// n = nt*16 + (l&15), k = kt*32 + ((l>=16)?8:0) + (j&7) + (j>=8?16:0); zero-pad k>=din.
__global__ void kn_swizzle(const float* __restrict__ w, int din, int ktiles,
                           int ntiles, __bf16* __restrict__ dst) {
  const int total = ktiles * ntiles * 512;
  for (int idx = blockIdx.x * blockDim.x + threadIdx.x; idx < total;
       idx += gridDim.x * blockDim.x) {
    const int tile = idx >> 9;
    const int e    = idx & 511;
    const int lane = e >> 4;
    const int j    = e & 15;
    const int nt   = tile / ktiles;
    const int kt   = tile - nt * ktiles;
    const int n    = nt * 16 + (lane & 15);
    const int k    = kt * 32 + ((lane >> 4) << 3) + (j & 7) + ((j >> 3) << 4);
    const float v  = (k < din) ? w[(size_t)n * din + k] : 0.f;
    dst[idx] = (__bf16)v;
  }
}

// ------------------------------ main kernel --------------------------------

__global__ __launch_bounds__(256)
void kn_step_kernel(KParams P) {
  __shared__ float sF[64], sH[64];
  __shared__ float s_bias[B_TOTAL];     // all biases staged in LDS
  __shared__ float s_xpost[128], s_xprev[128], s_xpriprev[128], s_yprev[128];
  __shared__ float s_xprior[128], s_dy[128];
  __shared__ float s_raw[4 * 128];
  __shared__ __bf16 s_fevol[16 * 32], s_fupd[16 * 32], s_fc7in[16 * 32];
  __shared__ __bf16 s_out5[16 * 160], s_out6[16 * 160], s_out7[16 * 320];
  __shared__ float s_hQ[1024], s_hSig[1024], s_hS[1024], s_hSigG[1024];
  __shared__ float s_gi[16 * 192], s_gh[16 * 192];
  __shared__ __bf16 s_cat[16 * 384];
  __shared__ __bf16 s_out1[1024], s_out3[1024];
  __shared__ float s_out2[1024];
  __shared__ __bf16 s_h2[8 * 1024];   // per-wave bf16 h2 chunk (16x64)
  __shared__ float s_big[8 * 1024];   // per-wave fc2b partial tiles

  const int tid  = threadIdx.x;
  const int lane = tid & 31;
  const int wave = tid >> 5;
  const int wg   = blockIdx.x;        // batch tile: rows wg*16 .. wg*16+15

  // ---- init carries / constants / biases / zero pads ----
  for (int e = tid; e < 64; e += 256) { sF[e] = P.F[e]; sH[e] = P.H[e]; }
  for (int e = tid; e < B_TOTAL; e += 256) s_bias[e] = P.biases[e];
  for (int e = tid; e < 128; e += 256) {
    s_xpost[e] = 0.f; s_xprev[e] = 0.f; s_xpriprev[e] = 0.f; s_yprev[e] = 0.f;
  }
  for (int e = tid; e < 1024; e += 256) {
    s_hQ[e] = 0.f; s_hSig[e] = 0.f; s_hS[e] = 0.f;
  }
  for (int e = tid; e < 16 * 32; e += 256) {
    s_fevol[e] = (__bf16)0.f; s_fupd[e] = (__bf16)0.f; s_fc7in[e] = (__bf16)0.f;
  }
  __syncthreads();

  for (int t = 0; t < 64; ++t) {
    // ---- A1: x_prior = x_post @ F^T ----
    if (tid < 128) {
      const int r = tid >> 3, c = tid & 7;
      float s = 0.f;
#pragma unroll
      for (int k = 0; k < 8; ++k) s += s_xpost[r * 8 + k] * sF[c * 8 + k];
      s_xprior[r * 8 + c] = s;
    }
    __syncthreads();

    // ---- A2: innovation + raw feature diffs ----
    if (tid < 128) {
      const int r = tid >> 3, c = tid & 7;
      float m1 = 0.f;
#pragma unroll
      for (int k = 0; k < 8; ++k) m1 += s_xprior[r * 8 + k] * sH[c * 8 + k];
      const int b = wg * 16 + r;
      const float yv = P.y[((size_t)b * 64 + t) * 8 + c];
      const float d  = yv - m1;
      s_dy[r * 8 + c] = d;
      s_raw[0 * 128 + r * 8 + c] = yv - s_yprev[r * 8 + c];
      s_raw[1 * 128 + r * 8 + c] = d;
      s_raw[2 * 128 + r * 8 + c] = s_xpost[r * 8 + c] - s_xprev[r * 8 + c];
      s_raw[3 * 128 + r * 8 + c] = s_xpost[r * 8 + c] - s_xpriprev[r * 8 + c];
    }
    __syncthreads();

    // ---- A3: L2-normalize rows -> bf16 A-buffers (K padded to 32) ----
    for (int e = tid; e < 512; e += 256) {
      const int f = e >> 7, r = (e >> 3) & 15, c = e & 7;
      const float* row = &s_raw[f * 128 + r * 8];
      float ss = 0.f;
#pragma unroll
      for (int k = 0; k < 8; ++k) ss += row[k] * row[k];
      const float v = row[c] / fmaxf(sqrtf(ss), 1e-12f);
      const __bf16 bv = (__bf16)v;
      if (f == 0)      s_fc7in[r * 32 + c]     = bv;   // obs_diff
      else if (f == 1) s_fc7in[r * 32 + 8 + c] = bv;   // obs_innov
      else if (f == 2) s_fevol[r * 32 + c]     = bv;   // fw_evol
      else             s_fupd[r * 32 + c]      = bv;   // fw_upd
    }
    __syncthreads();

    // ---- S1: fc5, fc6, fc7 (relu) ----
    gemm16(s_fevol, 32, P.sw + OFF_FC5, 1, 10, s_out5, 160, s_bias + B_FC5, true, wave, lane);
    gemm16(s_fupd,  32, P.sw + OFF_FC6, 1, 10, s_out6, 160, s_bias + B_FC6, true, wave, lane);
    gemm16(s_fc7in, 32, P.sw + OFF_FC7, 1, 20, s_out7, 320, s_bias + B_FC7, true, wave, lane);
    __syncthreads();

    // ---- S2: gruQ gi/gh ----
    gemm16(s_out5, 160, P.sw + OFF_Q_IH, 5, 12, s_gi, 192, s_bias + B_Q_IH, false, wave, lane);
    gemm16(s_hQ,    64, P.sw + OFF_Q_HH, 2, 12, s_gh, 192, s_bias + B_Q_HH, false, wave, lane);
    __syncthreads();

    // ---- S3: gruQ gates -> hQ (in place); build cat(hQ, out6) ----
    for (int e = tid; e < 1024; e += 256) {
      const int r = e >> 6, c = e & 63;
      const float rr = sigm(s_gi[r * 192 + c] + s_gh[r * 192 + c]);
      const float zz = sigm(s_gi[r * 192 + 64 + c] + s_gh[r * 192 + 64 + c]);
      const float gg = tanhf(s_gi[r * 192 + 128 + c] + rr * s_gh[r * 192 + 128 + c]);
      const float hn = (1.f - zz) * gg + zz * s_hQ[r * 64 + c];
      s_hQ[r * 64 + c] = hn;
      s_cat[r * 384 + c] = (__bf16)hn;
    }
    for (int e = tid; e < 16 * 160; e += 256) {
      const int r = e / 160, c = e - r * 160;
      s_cat[r * 384 + 64 + c] = s_out6[r * 160 + c];
    }
    __syncthreads();

    // ---- S4: gruSig gi/gh ----
    gemm16(s_cat, 384, P.sw + OFF_SIG_IH, 7, 12, s_gi, 192, s_bias + B_SIG_IH, false, wave, lane);
    gemm16(s_hSig, 64, P.sw + OFF_SIG_HH, 2, 12, s_gh, 192, s_bias + B_SIG_HH, false, wave, lane);
    __syncthreads();

    // ---- S5: gruSig gates -> hSigG ----
    for (int e = tid; e < 1024; e += 256) {
      const int r = e >> 6, c = e & 63;
      const float rr = sigm(s_gi[r * 192 + c] + s_gh[r * 192 + c]);
      const float zz = sigm(s_gi[r * 192 + 64 + c] + s_gh[r * 192 + 64 + c]);
      const float gg = tanhf(s_gi[r * 192 + 128 + c] + rr * s_gh[r * 192 + 128 + c]);
      s_hSigG[r * 64 + c] = (1.f - zz) * gg + zz * s_hSig[r * 64 + c];
    }
    __syncthreads();

    // ---- S6: fc1 (relu) ----
    gemm16(s_hSigG, 64, P.sw + OFF_FC1, 2, 4, s_out1, 64, s_bias + B_FC1, true, wave, lane);
    __syncthreads();

    // ---- S7: cat(out1, out7) ----
    for (int e = tid; e < 16 * 384; e += 256) {
      const int r = e / 384, c = e - r * 384;
      s_cat[e] = (c < 64) ? s_out1[r * 64 + c] : s_out7[r * 320 + (c - 64)];
    }
    __syncthreads();

    // ---- S8: gruS gi/gh ----
    gemm16(s_cat, 384, P.sw + OFF_S_IH, 12, 12, s_gi, 192, s_bias + B_S_IH, false, wave, lane);
    gemm16(s_hS,   64, P.sw + OFF_S_HH,  2, 12, s_gh, 192, s_bias + B_S_HH, false, wave, lane);
    __syncthreads();

    // ---- S9: gruS gates -> hS (in place); build cat2 = cat(hSigG, hS) ----
    for (int e = tid; e < 1024; e += 256) {
      const int r = e >> 6, c = e & 63;
      const float rr = sigm(s_gi[r * 192 + c] + s_gh[r * 192 + c]);
      const float zz = sigm(s_gi[r * 192 + 64 + c] + s_gh[r * 192 + 64 + c]);
      const float gg = tanhf(s_gi[r * 192 + 128 + c] + rr * s_gh[r * 192 + 128 + c]);
      const float hn = (1.f - zz) * gg + zz * s_hS[r * 64 + c];
      s_hS[r * 64 + c] = hn;
      s_cat[r * 384 + 64 + c] = (__bf16)hn;
      s_cat[r * 384 + c] = (__bf16)s_hSigG[r * 64 + c];
    }
    __syncthreads();

    // ---- S10: fused fc2a -> relu -> fc2b partials (wave-private K chunks) --
    {
      __bf16* slab = &s_h2[wave * 1024];          // 16x64 bf16 h2 chunk
      // hoist loop-invariant cat2 A-fragments (K = 4 tiles) into registers
      v16bf aA0 = load_a(s_cat, 384, 0, lane);
      v16bf aA1 = load_a(s_cat, 384, 1, lane);
      v16bf aA2 = load_a(s_cat, 384, 2, lane);
      v16bf aA3 = load_a(s_cat, 384, 3, lane);
      v8f p0 = zero8(), p1 = zero8(), p2 = zero8(), p3 = zero8();
      for (int ch = wave; ch < 80; ch += WAVES) { // chunk = 64 fc2a columns
        __builtin_prefetch(P.sw + OFF_FC2A + (size_t)(ch + WAVES) * 4 * 4 * 512, 0, 1);
#pragma unroll
        for (int i = 0; i < 4; ++i) {
          const int nt = ch * 4 + i;
          const __bf16* bp = P.sw + OFF_FC2A + (size_t)nt * (4u * 512u);
          v8f c = zero8();
          c = bfwmma(aA0, *(const v16bf*)(bp + 0 * 512), c);
          c = bfwmma(aA1, *(const v16bf*)(bp + 1 * 512), c);
          c = bfwmma(aA2, *(const v16bf*)(bp + 2 * 512), c);
          c = bfwmma(aA3, *(const v16bf*)(bp + 3 * 512), c);
          store_tile<__bf16>(slab, 64, i * 16, c,
                             s_bias + B_FC2A + nt * 16, true, lane);
        }
#pragma unroll
        for (int i = 0; i < 2; ++i) {
          v16bf av = load_a(slab, 64, i, lane);
          const int kt = ch * 2 + i;
          p0 = bfwmma(av, *(const v16bf*)(P.sw + OFF_FC2B + (size_t)(0 * 160 + kt) * 512), p0);
          p1 = bfwmma(av, *(const v16bf*)(P.sw + OFF_FC2B + (size_t)(1 * 160 + kt) * 512), p1);
          p2 = bfwmma(av, *(const v16bf*)(P.sw + OFF_FC2B + (size_t)(2 * 160 + kt) * 512), p2);
          p3 = bfwmma(av, *(const v16bf*)(P.sw + OFF_FC2B + (size_t)(3 * 160 + kt) * 512), p3);
        }
      }
      // write this wave's 4 partial tiles
      store_tile<float>(&s_big[wave * 1024 + 0 * 256], 16, 0, p0, nullptr, false, lane);
      store_tile<float>(&s_big[wave * 1024 + 1 * 256], 16, 0, p1, nullptr, false, lane);
      store_tile<float>(&s_big[wave * 1024 + 2 * 256], 16, 0, p2, nullptr, false, lane);
      store_tile<float>(&s_big[wave * 1024 + 3 * 256], 16, 0, p3, nullptr, false, lane);
    }
    __syncthreads();

    // ---- S11: reduce fc2b partials -> out2; build cat3 = cat(hS, out2) ----
    for (int e = tid; e < 1024; e += 256) {
      const int r = e >> 6, c = e & 63;
      const int ntile = c >> 4, cc = c & 15;
      float s = s_bias[B_FC2B + c];
#pragma unroll
      for (int w = 0; w < 8; ++w) s += s_big[w * 1024 + ntile * 256 + r * 16 + cc];
      s_out2[r * 64 + c] = s;
      s_cat[r * 384 + 64 + c] = (__bf16)s;
      s_cat[r * 384 + c] = (__bf16)s_hS[r * 64 + c];
    }
    __syncthreads();

    // ---- S12: fc3 (relu) ----
    gemm16(s_cat, 384, P.sw + OFF_FC3, 4, 4, s_out3, 64, s_bias + B_FC3, true, wave, lane);
    __syncthreads();

    // ---- S13: cat4 = cat(hSigG, out3) ----
    for (int e = tid; e < 1024; e += 256) {
      const int r = e >> 6, c = e & 63;
      s_cat[r * 384 + c] = (__bf16)s_hSigG[r * 64 + c];
      s_cat[r * 384 + 64 + c] = s_out3[r * 64 + c];
    }
    __syncthreads();

    // ---- S14: fc4 (relu) -> next hSig carry (fp32) ----
    gemm16(s_cat, 384, P.sw + OFF_FC4, 4, 4, s_hSig, 64, s_bias + B_FC4, true, wave, lane);
    __syncthreads();

    // ---- S15: x_post' = x_prior + KG @ dy; write output; rotate carries ----
    if (tid < 128) {
      const int r = tid >> 3, m = tid & 7;
      float s = s_xprior[r * 8 + m];
#pragma unroll
      for (int n = 0; n < 8; ++n) s += s_out2[r * 64 + m * 8 + n] * s_dy[r * 8 + n];
      const int b = wg * 16 + r;
      P.out[((size_t)b * 64 + t) * 8 + m] = s;
      s_xprev[r * 8 + m]    = s_xpost[r * 8 + m];
      s_xpost[r * 8 + m]    = s;
      s_xpriprev[r * 8 + m] = s_xprior[r * 8 + m];
      s_yprev[r * 8 + m]    = P.y[((size_t)b * 64 + t) * 8 + m];
    }
    __syncthreads();
  }
}

// ------------------------------- host side ---------------------------------

extern "C" void kernel_launch(void* const* d_in, const int* in_sizes, int n_in,
                              void* d_out, int out_size, void* d_ws, size_t ws_size,
                              hipStream_t stream) {
  (void)in_sizes; (void)n_in; (void)out_size;
  if (ws_size < WS_TOTAL_BYTES) return;

  __bf16* sw = (__bf16*)d_ws;
  float*  bias_blk = (float*)((char*)d_ws + WS_BIAS_BYTE_OFF);

  struct SwJob { int in_idx, din, ktiles, ntiles; unsigned off; };
  static const SwJob jobs[14] = {
    { 1, 160,   5,  12, OFF_Q_IH   },  // gruQ_wih
    { 2,  64,   2,  12, OFF_Q_HH   },  // gruQ_whh
    { 5, 224,   7,  12, OFF_SIG_IH },  // gruSig_wih
    { 6,  64,   2,  12, OFF_SIG_HH },  // gruSig_whh
    { 9, 384,  12,  12, OFF_S_IH   },  // gruS_wih
    {10,  64,   2,  12, OFF_S_HH   },  // gruS_whh
    {13,  64,   2,   4, OFF_FC1    },  // fc1_w
    {15, 128,   4, 320, OFF_FC2A   },  // fc2a_w
    {17, 5120, 160,  4, OFF_FC2B   },  // fc2b_w
    {19, 128,   4,   4, OFF_FC3    },  // fc3_w
    {21, 128,   4,   4, OFF_FC4    },  // fc4_w
    {23,   8,   1,  10, OFF_FC5    },  // fc5_w (K padded 8->32)
    {25,   8,   1,  10, OFF_FC6    },  // fc6_w
    {27,  16,   1,  20, OFF_FC7    },  // fc7_w (K padded 16->32)
  };
  for (int i = 0; i < 14; ++i) {
    const int total = jobs[i].ktiles * jobs[i].ntiles * 512;
    const int blocks = (total + 255) / 256;
    kn_swizzle<<<blocks, 256, 0, stream>>>((const float*)d_in[jobs[i].in_idx],
                                           jobs[i].din, jobs[i].ktiles,
                                           jobs[i].ntiles, sw + jobs[i].off);
  }

  // pack biases contiguously into d_ws (device-to-device, capture-safe)
  struct BJob { int in_idx, nelem, off; };
  static const BJob bjobs[14] = {
    { 3, 192,  B_Q_IH   }, { 4, 192,  B_Q_HH   },
    { 7, 192,  B_SIG_IH }, { 8, 192,  B_SIG_HH },
    {11, 192,  B_S_IH   }, {12, 192,  B_S_HH   },
    {14,  64,  B_FC1    }, {16, 5120, B_FC2A   },
    {18,  64,  B_FC2B   }, {20,  64,  B_FC3    },
    {22,  64,  B_FC4    }, {24, 160,  B_FC5    },
    {26, 160,  B_FC6    }, {28, 320,  B_FC7    },
  };
  for (int i = 0; i < 14; ++i) {
    hipMemcpyAsync(bias_blk + bjobs[i].off, d_in[bjobs[i].in_idx],
                   (size_t)bjobs[i].nelem * sizeof(float),
                   hipMemcpyDeviceToDevice, stream);
  }

  KParams P;
  P.y      = (const float*)d_in[0];
  P.biases = bias_blk;
  P.F      = (const float*)d_in[29];
  P.H      = (const float*)d_in[30];
  P.sw     = sw;
  P.out    = (float*)d_out;

  kn_step_kernel<<<16, 256, 0, stream>>>(P);
}
